// ConceptEncodingBlock_2241972929153
// MI455X (gfx1250) — matrix operand order repaired and checked by reading, not compiled, not run on hardware
//
#include <hip/hip_runtime.h>

// Problem constants (from reference)
#define BB   4
#define LL   512
#define MM   32
#define EMBD 512
#define HH   8
#define HSZ  64
#define BL   (BB * LL)

typedef __attribute__((ext_vector_type(16))) _Float16 v16h;
typedef __attribute__((ext_vector_type(8)))  _Float16 v8h;
typedef __attribute__((ext_vector_type(8)))  float    v8f;
typedef __attribute__((ext_vector_type(4)))  float    v4f;

// ---------------------------------------------------------------------------
// WMMA helpers (gfx1250 wave32, 16x16x32 f16 -> f32).
// Operand layouts used here (contiguous K, same for A and B; not ISA table order):
//  A (16x32, f16): M = lane&15, elems 0..15 = K off..off+15 (off = lane<16 ? 0 : 16).
//  B (32x16, f16): N = lane&15, elems 0..15 = K off..off+15 (off = lane<16 ? 0 : 16).
//  C/D (16x16,f32): N = lane&15, row M = vgpr + (lane<16 ? 0 : 8).
// ---------------------------------------------------------------------------
__device__ __forceinline__ v8f wmma_f16(v16h a, v16h b, v8f c) {
  return __builtin_amdgcn_wmma_f32_16x16x32_f16(false, a, false, b, (short)0, c,
                                                false, false);
}

// rowPtr = &A[row * ld + kb]  (row-major, K contiguous)
__device__ __forceinline__ v16h gather_a(const _Float16* rowPtr, int lane) {
  const int off = (lane < 16) ? 0 : 16;
  v8h lo = *(const v8h*)(rowPtr + off);
  v8h hi = *(const v8h*)(rowPtr + off + 8);
  v16h a;
#pragma unroll
  for (int e = 0; e < 8; ++e) { a[e] = lo[e]; a[8 + e] = hi[e]; }
  return a;
}

// colPtr = &B[n * ld + kb]  where B is stored [N][K] row-major (K contiguous)
__device__ __forceinline__ v16h gather_b(const _Float16* colPtr, int lane) {
  const int off = (lane < 16) ? 0 : 16;
  v8h lo = *(const v8h*)(colPtr + off);
  v8h hi = *(const v8h*)(colPtr + off + 8);
  v16h b;
#pragma unroll
  for (int e = 0; e < 8; ++e) { b[e] = lo[e]; b[8 + e] = hi[e]; }
  return b;
}

// Same as gather_b but converts from f32 memory (used for v, read exactly once)
__device__ __forceinline__ v16h gather_b_f32(const float* colPtr, int lane) {
  const int off = (lane < 16) ? 0 : 16;
  v16h b;
#pragma unroll
  for (int q = 0; q < 4; ++q) {
    v4f f = *(const v4f*)(colPtr + off + q * 4);
#pragma unroll
    for (int e = 0; e < 4; ++e) b[q * 4 + e] = (_Float16)f[e];
  }
  return b;
}

// ---------------------------------------------------------------------------
// K0: f32 -> f16 cast (q_w, cells)
// ---------------------------------------------------------------------------
__global__ void __launch_bounds__(256)
cast_kernel(const float* __restrict__ src, _Float16* __restrict__ dst, int n) {
  int i = blockIdx.x * blockDim.x + threadIdx.x;
  int stride = gridDim.x * blockDim.x;
  for (; i < n; i += stride) dst[i] = (_Float16)src[i];
}

// ---------------------------------------------------------------------------
// K1: LayerNorm; writes h16 [bl][emb] and transposed hT16 [b][emb][l]
// One 256-thread block per row (512 elems, 2/thread).
// ---------------------------------------------------------------------------
__global__ void __launch_bounds__(256)
layernorm_kernel(const float* __restrict__ x, const float* __restrict__ ln_g,
                 const float* __restrict__ ln_b, _Float16* __restrict__ h16,
                 _Float16* __restrict__ hT16) {
  const int row = blockIdx.x;           // 0..BL-1
  const int b = row >> 9, l = row & 511;
  const float* xr = x + (size_t)row * EMBD;
  const int t = threadIdx.x;
  float x0 = xr[t], x1 = xr[t + 256];
  __shared__ float r1[256], r2[256];
  r1[t] = x0 + x1;
  r2[t] = x0 * x0 + x1 * x1;
  __syncthreads();
  for (int s = 128; s > 0; s >>= 1) {
    if (t < s) { r1[t] += r1[t + s]; r2[t] += r2[t + s]; }
    __syncthreads();
  }
  float mu = r1[0] * (1.0f / EMBD);
  float var = r2[0] * (1.0f / EMBD) - mu * mu;
  float rstd = rsqrtf(var + 1e-5f);
  float y0 = (x0 - mu) * rstd * ln_g[t] + ln_b[t];
  float y1 = (x1 - mu) * rstd * ln_g[t + 256] + ln_b[t + 256];
  h16[(size_t)row * EMBD + t] = (_Float16)y0;
  h16[(size_t)row * EMBD + t + 256] = (_Float16)y1;
  hT16[((size_t)b * EMBD + t) * LL + l] = (_Float16)y0;
  hT16[((size_t)b * EMBD + t + 256) * LL + l] = (_Float16)y1;
}

// ---------------------------------------------------------------------------
// K2: q = h @ q_w^T + q_b   (2048 x 512 x 512), one wave per 16x16 tile.
// Writes q16 in [b][h][l][s] layout for the scores GEMM.
// ---------------------------------------------------------------------------
__global__ void __launch_bounds__(256)
qproj_kernel(const _Float16* __restrict__ h16, const _Float16* __restrict__ qw16,
             const float* __restrict__ q_b, _Float16* __restrict__ q16) {
  const int lane = threadIdx.x & 31;
  const int tile = blockIdx.x * (blockDim.x >> 5) + (threadIdx.x >> 5); // 0..4095
  const int rowBase = (tile >> 5) << 4;  // over BL
  const int colBase = (tile & 31) << 4;  // over EMB
  const int arow = rowBase + (lane & 15);
  const int bcol = colBase + (lane & 15);
  v8f acc = {};
  for (int kb = 0; kb < EMBD; kb += 32) {
    v16h a = gather_a(h16 + (size_t)arow * EMBD + kb, lane);
    v16h bm = gather_b(qw16 + (size_t)bcol * EMBD + kb, lane);
    acc = wmma_f16(a, bm, acc);
  }
  const int j = colBase + (lane & 15);
  const int hh = j >> 6, ss = j & 63;
  const float bias = q_b[j];
  const int rhi = (lane < 16) ? 0 : 8;
#pragma unroll
  for (int r = 0; r < 8; ++r) {
    int i = rowBase + r + rhi;
    int b_ = i >> 9, l_ = i & 511;
    q16[(((size_t)(b_ * HH + hh) * LL + l_) * HSZ) + ss] = (_Float16)(acc[r] + bias);
  }
}

// ---------------------------------------------------------------------------
// K3: scores[b,h,m,l] = (q[b,h,l,:] . cells[m,h,:]) * HS^-0.5
// Per (b,h): (512 x 64) @ (64 x 32). 64 tiles per bh * 32 bh = 2048 waves.
// ---------------------------------------------------------------------------
__global__ void __launch_bounds__(256)
scores_kernel(const _Float16* __restrict__ q16, const _Float16* __restrict__ cells16,
              float* __restrict__ scores) {
  const int lane = threadIdx.x & 31;
  const int tile = blockIdx.x * (blockDim.x >> 5) + (threadIdx.x >> 5); // 0..2047
  const int bh = tile >> 6;
  const int t = tile & 63;
  const int rowBase = (t >> 1) << 4;  // l
  const int colBase = (t & 1) << 4;   // m
  const int h = bh & 7;
  const int arow = rowBase + (lane & 15);
  const int bm = colBase + (lane & 15);
  v8f acc = {};
  for (int kb = 0; kb < HSZ; kb += 32) {
    v16h a = gather_a(q16 + ((size_t)bh * LL + arow) * HSZ + kb, lane);
    v16h bmv = gather_b(cells16 + (size_t)bm * (HH * HSZ) + h * HSZ + kb, lane);
    acc = wmma_f16(a, bmv, acc);
  }
  const int rhi = (lane < 16) ? 0 : 8;
#pragma unroll
  for (int r = 0; r < 8; ++r) {
    int l = rowBase + r + rhi;
    scores[((size_t)bh * MM + bm) * LL + l] = acc[r] * 0.125f;
  }
}

// ---------------------------------------------------------------------------
// K4: softmax over l (512) per (b,h,m) row; writes w16 in [b][m*8+h][l]
// layout so it is the A operand of the g GEMM.
// ---------------------------------------------------------------------------
__global__ void __launch_bounds__(256)
softmax_kernel(const float* __restrict__ scores, _Float16* __restrict__ w16) {
  const int row = blockIdx.x;  // ((b*8+h)*32+m)
  const int b = row >> 8;
  const int h = (row >> 5) & 7;
  const int m = row & 31;
  const float* src = scores + (size_t)row * LL;
  _Float16* dst = w16 + ((size_t)b * (MM * HH) + m * HH + h) * LL;
  __shared__ float red[256];
  const int t = threadIdx.x;
  float x0 = src[t], x1 = src[t + 256];
  red[t] = fmaxf(x0, x1);
  __syncthreads();
  for (int s = 128; s > 0; s >>= 1) {
    if (t < s) red[t] = fmaxf(red[t], red[t + s]);
    __syncthreads();
  }
  float mx = red[0];
  __syncthreads();
  float e0 = __expf(x0 - mx), e1 = __expf(x1 - mx);
  red[t] = e0 + e1;
  __syncthreads();
  for (int s = 128; s > 0; s >>= 1) {
    if (t < s) red[t] += red[t + s];
    __syncthreads();
  }
  float inv = 1.0f / red[0];
  dst[t] = (_Float16)(e0 * inv);
  dst[t + 256] = (_Float16)(e1 * inv);
}

// ---------------------------------------------------------------------------
// K5: g[b, m*8+h, vdim] = sum_l w[b,h,m,l] * h[b,l,vdim]
// Per b: (256 x 512) @ (512 x 512) with B = hT16 (K-contiguous). 2048 waves.
// ---------------------------------------------------------------------------
__global__ void __launch_bounds__(256)
g_kernel(const _Float16* __restrict__ w16, const _Float16* __restrict__ hT16,
         _Float16* __restrict__ g16) {
  const int lane = threadIdx.x & 31;
  const int tile = blockIdx.x * (blockDim.x >> 5) + (threadIdx.x >> 5); // 0..2047
  const int b = tile >> 9;
  const int t = tile & 511;
  const int rowBase = (t >> 5) << 4;  // over 256 rows (m*8+h)
  const int colBase = (t & 31) << 4;  // over EMB
  const int arow = rowBase + (lane & 15);
  const int bcol = colBase + (lane & 15);
  const _Float16* wrow = w16 + ((size_t)b * (MM * HH) + arow) * LL;
  const _Float16* hcol = hT16 + ((size_t)b * EMBD + bcol) * LL;
  v8f acc = {};
  for (int kb = 0; kb < LL; kb += 32) {
    v16h a = gather_a(wrow + kb, lane);
    v16h bm = gather_b(hcol + kb, lane);
    acc = wmma_f16(a, bm, acc);
  }
  const int rhi = (lane < 16) ? 0 : 8;
#pragma unroll
  for (int r = 0; r < 8; ++r) {
    int rg = rowBase + r + rhi;  // m*8+h
    g16[((size_t)b * (MM * HH) + rg) * EMBD + bcol] = (_Float16)acc[r];
  }
}

// ---------------------------------------------------------------------------
// K6: out[b, h*2048 + m*64 + s] = sum_v v[m, h*64+s, v] * g[b,m,h,v] + vb[m,h*64+s]
// Per (m,h): (4pad16 x 512) @ (512 x 64). v converted f32->f16 in-register
// (read from HBM exactly once). 1024 waves.
// ---------------------------------------------------------------------------
__global__ void __launch_bounds__(256)
out_kernel(const _Float16* __restrict__ g16, const float* __restrict__ v,
           const float* __restrict__ vb, float* __restrict__ out) {
  const int lane = threadIdx.x & 31;
  const int tile = blockIdx.x * (blockDim.x >> 5) + (threadIdx.x >> 5); // 0..1023
  const int mh = tile >> 2;  // m*8+h
  const int m = mh >> 3, h = mh & 7;
  const int colBase = (tile & 3) << 4;    // s within 64
  const int bi = lane & 15;               // padded batch row
  const int n = colBase + (lane & 15);    // s local
  const int j = h * HSZ + n;              // column of v[m]
  const int bclamp = (bi < 4) ? bi : 3;
  const _Float16 amask = (bi < 4) ? (_Float16)1.0f : (_Float16)0.0f;
  const _Float16* arow = g16 + ((size_t)bclamp * (MM * HH) + mh) * EMBD;
  const float* brow = v + ((size_t)m * EMBD + j) * EMBD;
  v8f acc = {};
  for (int kb = 0; kb < EMBD; kb += 32) {
    v16h a = gather_a(arow + kb, lane);
#pragma unroll
    for (int e = 0; e < 16; ++e) a[e] *= amask;
    v16h bm = gather_b_f32(brow + kb, lane);
    acc = wmma_f16(a, bm, acc);
  }
  if (lane < 16) {
    const float bias = vb[(size_t)m * EMBD + j];
    const int flat = h * (MM * HSZ) + m * HSZ + n;
#pragma unroll
    for (int r = 0; r < 4; ++r)  // rows 0..3 = batch
      out[(size_t)r * (MM * EMBD) + flat] = acc[r] + bias;
  }
}

// ---------------------------------------------------------------------------
extern "C" void kernel_launch(void* const* d_in, const int* in_sizes, int n_in,
                              void* d_out, int out_size, void* d_ws, size_t ws_size,
                              hipStream_t stream) {
  (void)in_sizes; (void)n_in; (void)out_size; (void)ws_size;
  const float* x     = (const float*)d_in[0];
  const float* cells = (const float*)d_in[1];
  const float* q_w   = (const float*)d_in[2];
  const float* q_b   = (const float*)d_in[3];
  const float* v     = (const float*)d_in[4];
  const float* vb    = (const float*)d_in[5];
  const float* ln_g  = (const float*)d_in[6];
  const float* ln_b  = (const float*)d_in[7];
  float* out = (float*)d_out;

  char* base = (char*)d_ws;
  size_t off = 0;
  _Float16* h16     = (_Float16*)(base + off); off += (size_t)BL * EMBD * 2;        // 2 MB
  _Float16* hT16    = (_Float16*)(base + off); off += (size_t)BB * EMBD * LL * 2;   // 2 MB
  _Float16* qw16    = (_Float16*)(base + off); off += (size_t)EMBD * EMBD * 2;      // 0.5 MB
  _Float16* cells16 = (_Float16*)(base + off); off += (size_t)MM * HH * HSZ * 2;    // 32 KB
  _Float16* q16     = (_Float16*)(base + off); off += (size_t)BB * HH * LL * HSZ * 2; // 2 MB
  float*    scores  = (float*)(base + off);    off += (size_t)BB * HH * MM * LL * 4;  // 2 MB
  _Float16* w16     = (_Float16*)(base + off); off += (size_t)BB * MM * HH * LL * 2;  // 1 MB
  _Float16* g16     = (_Float16*)(base + off); off += (size_t)BB * MM * HH * EMBD * 2;// 1 MB

  cast_kernel<<<(EMBD * EMBD + 255) / 256, 256, 0, stream>>>(q_w, qw16, EMBD * EMBD);
  cast_kernel<<<(MM * HH * HSZ + 255) / 256, 256, 0, stream>>>(cells, cells16, MM * HH * HSZ);
  layernorm_kernel<<<BL, 256, 0, stream>>>(x, ln_g, ln_b, h16, hT16);
  qproj_kernel<<<512, 256, 0, stream>>>(h16, qw16, q_b, q16);     // 4096 tiles
  scores_kernel<<<256, 256, 0, stream>>>(q16, cells16, scores);   // 2048 tiles
  softmax_kernel<<<BB * HH * MM, 256, 0, stream>>>(scores, w16);  // 1024 rows
  g_kernel<<<256, 256, 0, stream>>>(w16, hT16, g16);              // 2048 tiles
  out_kernel<<<128, 256, 0, stream>>>(g16, v, vb, out);           // 1024 tiles
}
